// IACRMoEBlock_80118319939665
// MI455X (gfx1250) — matrix operand order repaired and checked by reading, not compiled, not run on hardware
//
#include <hip/hip_runtime.h>
#include <hip/hip_bf16.h>

// ---------------------------------------------------------------------------
// IACRMoEBlock for MI455X (gfx1250): bf16 WMMA for the dense 11-expert MLP,
// TDM (tensor_load_to_lds) weight staging shared by all waves of a block,
// branchless gelu, scalar kernels for the tiny attention/routing parts.
// ---------------------------------------------------------------------------

typedef __bf16 bf16;
typedef __attribute__((ext_vector_type(16))) bf16  v16bf;
typedef __attribute__((ext_vector_type(8)))  bf16  v8bf;
typedef __attribute__((ext_vector_type(8)))  float v8f;
typedef __attribute__((ext_vector_type(4)))  unsigned int u32x4;
typedef __attribute__((ext_vector_type(8)))  int i32x8;
typedef __attribute__((ext_vector_type(4)))  int i32x4;

#define B_   8
#define C_   128
#define HW_  4096
#define P_   64
#define E_   11
#define HID_ 512
#define NPIX (B_ * HW_)   // 32768

__device__ __forceinline__ bf16 f2bf(float f) {
    union { float f; unsigned u; } v; v.f = f;
    unsigned r = v.u + 0x7fffu + ((v.u >> 16) & 1u);   // round-to-nearest-even
    unsigned short h = (unsigned short)(r >> 16);
    return __builtin_bit_cast(bf16, h);
}

// Branchless erf (Abramowitz-Stegun 7.1.26, |err| < 1.5e-7):
// exp -> v_exp_f32, reciprocal -> v_rcp_f32 (no IEEE-div fixup chain).
__device__ __forceinline__ float erf_fast(float z) {
    float a  = fabsf(z);
    float t  = __builtin_amdgcn_rcpf(1.0f + 0.3275911f * a);
    float p  = t * (0.254829592f + t * (-0.284496736f + t * (1.421413741f +
               t * (-1.453152027f + t * 1.061405429f))));
    float e  = 1.0f - p * __expf(-a * a);
    return copysignf(e, z);
}
__device__ __forceinline__ float gelu_exact(float v) {
    return 0.5f * v * (1.0f + erf_fast(v * 0.70710678118f));
}

// generic pointer -> 32-bit LDS byte offset (for TDM descriptors)
__device__ __forceinline__ unsigned lds_addr32(const void* p) {
    return (unsigned)(unsigned long long)(__attribute__((address_space(3))) const char*)p;
}

// ---------------------------------------------------------------------------
// TDM: DMA a 2D bf16 tile (tile_w x tile_h elems, row stride in elems) into LDS.
// D# group0: count=1, lds_addr, global_addr, type=2 ("image").
// D# group1: data_size=2B, generous tensor dims (no clipping), tile dims, stride.
// This toolchain's builtin takes 6 args (g0, g1, g2, g3, extra group, cpol).
__device__ __forceinline__ void tdm_load_2d(unsigned lds_byte, const void* gptr,
                                            unsigned tile_w, unsigned tile_h,
                                            unsigned long long row_stride_elems) {
    unsigned long long ga = (unsigned long long)gptr;
    const unsigned td0 = 1u << 20, td1 = 1u << 20;    // tensor bounds >> tile
    u32x4 g0;
    g0[0] = 1u;                                        // count = 1 valid D#
    g0[1] = lds_byte;                                  // LDS destination (bytes)
    g0[2] = (unsigned)ga;                              // global_addr[31:0]
    g0[3] = (unsigned)((ga >> 32) & 0x01ffffffu) | (2u << 30);  // [56:32] | type=2
    i32x8 g1;
    g1[0] = (int)(1u << 16);                           // wgmask=0, data_size=2B
    g1[1] = (int)((td0 & 0xffffu) << 16);              // tensor_dim0[15:0]
    g1[2] = (int)(((td0 >> 16) & 0xffffu) | ((td1 & 0xffffu) << 16));
    g1[3] = (int)(((td1 >> 16) & 0xffffu) | ((tile_w & 0xffffu) << 16));
    g1[4] = (int)(tile_h & 0xffffu);                   // tile_dim1 ; tile_dim2=0
    g1[5] = (int)(unsigned)(row_stride_elems & 0xffffffffu);   // dim0 stride lo
    g1[6] = (int)(unsigned)((row_stride_elems >> 32) & 0xffffu);
    g1[7] = 0;
    i32x4 g2 = {0, 0, 0, 0}, g3 = {0, 0, 0, 0};
    i32x8 g4 = {0, 0, 0, 0, 0, 0, 0, 0};
    __builtin_amdgcn_tensor_load_to_lds(g0, g1, g2, g3, g4, 0);
}

// ---------------------------------------------------------------------------
__global__ void k_zero(float* gprob, float* gload) {
    int t = threadIdx.x;
    if (t < E_) { gprob[t] = 0.0f; gload[t] = 0.0f; }
}

// expert weights f32 -> bf16 (both row-major, K contiguous)
__global__ void k_wcvt(const float* __restrict__ ew1, const float* __restrict__ ew2,
                       bf16* __restrict__ w1, bf16* __restrict__ w2, int n) {
    int i = blockIdx.x * 256 + threadIdx.x;
    if (i < n) { w1[i] = f2bf(ew1[i]); w2[i] = f2bf(ew2[i]); }
}

// x (b, c, pix) f32 -> xbf (b, pix, c) bf16, 16x16 LDS tile transpose
__global__ __launch_bounds__(256) void k_transpose(const float* __restrict__ x,
                                                   bf16* __restrict__ xbf) {
    __shared__ bf16 tile[16][17];
    int b  = blockIdx.z;
    int c0 = blockIdx.y * 16;
    int p0 = blockIdx.x * 16;
    int tx = threadIdx.x, ty = threadIdx.y;
    float v = x[((size_t)(b * C_ + c0 + ty)) * HW_ + p0 + tx];
    tile[ty][tx] = f2bf(v);
    __syncthreads();
    xbf[((size_t)b * HW_ + p0 + ty) * C_ + c0 + tx] = tile[tx][ty];
}

// ---------------------------------------------------------------------------
// per-batch: gc -> ctx proj -> [gc; proto] -> MHA(S=12,P=64,NH=4) -> LN -> up
__global__ __launch_bounds__(64) void k_prep(
    const float* __restrict__ x,   const float* __restrict__ proto,
    const float* __restrict__ ctx_w, const float* __restrict__ ctx_b,
    const float* __restrict__ wq, const float* __restrict__ bq,
    const float* __restrict__ wk, const float* __restrict__ bk,
    const float* __restrict__ wv, const float* __restrict__ bv,
    const float* __restrict__ wo, const float* __restrict__ bo,
    const float* __restrict__ ln_g, const float* __restrict__ ln_b,
    float* __restrict__ up, float* __restrict__ ortho_sq)
{
    int b = blockIdx.x;
    int t = threadIdx.x;                       // 64 threads
    __shared__ float gc[C_];
    __shared__ float seq[12][P_];
    __shared__ float qm[12][P_], km[12][P_], vm[12][P_];
    __shared__ float ao[12][P_];
    __shared__ float red[64];

    for (int c = 0; c < C_; ++c) {
        const float* xp = x + ((size_t)(b * C_ + c)) * HW_;
        float s = 0.f;
        for (int i = t; i < HW_; i += 64) s += xp[i];
        red[t] = s; __syncthreads();
        for (int off = 32; off > 0; off >>= 1) { if (t < off) red[t] += red[t + off]; __syncthreads(); }
        if (t == 0) gc[c] = red[0] * (1.0f / (float)HW_);
        __syncthreads();
    }
    {
        float s = ctx_b[t];
        for (int c = 0; c < C_; ++c) s += ctx_w[t * C_ + c] * gc[c];
        seq[0][t] = s;
    }
    for (int e = 0; e < E_; ++e) seq[1 + e][t] = proto[e * P_ + t];
    __syncthreads();
    for (int s = 0; s < 12; ++s) {
        float aq = bq[t], ak = bk[t], av = bv[t];
        for (int c = 0; c < P_; ++c) {
            float sv = seq[s][c];
            aq = fmaf(wq[t * P_ + c], sv, aq);
            ak = fmaf(wk[t * P_ + c], sv, ak);
            av = fmaf(wv[t * P_ + c], sv, av);
        }
        qm[s][t] = aq; km[s][t] = ak; vm[s][t] = av;
    }
    __syncthreads();
    if (t < 48) {                              // one thread per (s, h)
        int s = t % 12, h = t / 12;
        float sc[12], mx = -1e30f;
        for (int j = 0; j < 12; ++j) {
            float d = 0.f;
            for (int dd = 0; dd < 16; ++dd) d += qm[s][h * 16 + dd] * km[j][h * 16 + dd];
            sc[j] = d * 0.25f;
            mx = fmaxf(mx, sc[j]);
        }
        float ssum = 0.f;
        for (int j = 0; j < 12; ++j) { sc[j] = __expf(sc[j] - mx); ssum += sc[j]; }
        float inv = 1.0f / ssum;
        for (int dd = 0; dd < 16; ++dd) {
            float o = 0.f;
            for (int j = 0; j < 12; ++j) o += sc[j] * inv * vm[j][h * 16 + dd];
            ao[s][h * 16 + dd] = o;
        }
    }
    __syncthreads();
    for (int s = 0; s < 12; ++s) {
        float o = bo[t];
        for (int c = 0; c < P_; ++c) o = fmaf(wo[t * P_ + c], ao[s][c], o);
        qm[s][t] = o + seq[s][t];
    }
    __syncthreads();
    for (int s = 0; s < 12; ++s) {
        red[t] = qm[s][t]; __syncthreads();
        for (int off = 32; off > 0; off >>= 1) { if (t < off) red[t] += red[t + off]; __syncthreads(); }
        float mean = red[0] * (1.0f / 64.0f); __syncthreads();
        float d = qm[s][t] - mean;
        red[t] = d * d; __syncthreads();
        for (int off = 32; off > 0; off >>= 1) { if (t < off) red[t] += red[t + off]; __syncthreads(); }
        float var = red[0] * (1.0f / 64.0f); __syncthreads();
        float y = d * rsqrtf(var + 1e-5f) * ln_g[t] + ln_b[t];
        if (s >= 1) up[((size_t)b * E_ + (s - 1)) * P_ + t] = y;
        __syncthreads();
    }
    if (b == 0) {                              // proto ortho penalty (squared)
        __shared__ float nrm[E_];
        if (t < E_) {
            float s = 0.f;
            for (int c = 0; c < P_; ++c) { float v = proto[t * P_ + c]; s += v * v; }
            nrm[t] = rsqrtf(s);
        }
        __syncthreads();
        float acc = 0.f;
        for (int idx = t; idx < E_ * E_; idx += 64) {
            int i = idx / E_, j = idx % E_;
            float d = 0.f;
            for (int c = 0; c < P_; ++c) d += proto[i * P_ + c] * proto[j * P_ + c];
            d *= nrm[i] * nrm[j];
            if (i == j) d -= 1.0f;
            acc += d * d;
        }
        red[t] = acc; __syncthreads();
        for (int off = 32; off > 0; off >>= 1) { if (t < off) red[t] += red[t + off]; __syncthreads(); }
        if (t == 0) *ortho_sq = red[0];
    }
}

// ---------------------------------------------------------------------------
// routing: xr = inp_w @ x + inp_b ; logits vs up ; softmax ; top-6 mask ; aux sums
__global__ __launch_bounds__(256) void k_route(
    const float* __restrict__ x, const float* __restrict__ inp_w,
    const float* __restrict__ inp_b, const float* __restrict__ up,
    float* __restrict__ mask, float* __restrict__ gprob, float* __restrict__ gload)
{
    __shared__ float wl[P_ * C_];          // 32 KB
    __shared__ float ul[E_ * P_];
    __shared__ float sprob[E_], sload[E_];
    int t   = threadIdx.x;
    int b   = blockIdx.x >> 4;
    int pix = ((blockIdx.x & 15) << 8) + t;

    for (int i = t; i < P_ * C_; i += 256) wl[i] = inp_w[i];
    for (int i = t; i < E_ * P_; i += 256) ul[i] = up[b * E_ * P_ + i];
    if (t < E_) { sprob[t] = 0.f; sload[t] = 0.f; }
    __syncthreads();

    float xr[P_];
    #pragma unroll
    for (int j = 0; j < P_; ++j) xr[j] = inp_b[j];
    for (int c = 0; c < C_; ++c) {
        float xc = x[((size_t)(b * C_ + c)) * HW_ + pix];
        #pragma unroll
        for (int j = 0; j < P_; ++j) xr[j] = fmaf(wl[j * C_ + c], xc, xr[j]);
    }
    float lg[E_]; float mx = -1e30f;
    for (int e = 0; e < E_; ++e) {
        float d = 0.f;
        #pragma unroll
        for (int j = 0; j < P_; ++j) d += ul[e * P_ + j] * xr[j];
        lg[e] = d * 0.125f;
        mx = fmaxf(mx, lg[e]);
    }
    float ssum = 0.f;
    for (int e = 0; e < E_; ++e) { lg[e] = __expf(lg[e] - mx); ssum += lg[e]; }
    float inv = 1.0f / ssum;
    bool sel[E_];
    for (int e = 0; e < E_; ++e) { lg[e] *= inv; sel[e] = false; }
    float tsum = 0.f;
    for (int k = 0; k < 6; ++k) {
        int bi = 0; float bv = -1.f;
        for (int e = 0; e < E_; ++e) if (!sel[e] && lg[e] > bv) { bv = lg[e]; bi = e; }
        sel[bi] = true; tsum += bv;
    }
    float tinv = 1.0f / tsum;
    for (int e = 0; e < E_; ++e) {
        mask[((size_t)(b * E_ + e)) * HW_ + pix] = sel[e] ? lg[e] * tinv : 0.0f;
        atomicAdd(&sprob[e], lg[e]);
        if (sel[e]) atomicAdd(&sload[e], 1.0f);
    }
    __syncthreads();
    if (t < E_) { atomicAdd(&gprob[t], sprob[t]); atomicAdd(&gload[t], sload[t]); }
}

// ---------------------------------------------------------------------------
// WMMA A-fragment: row-major [M x K] bf16, K contiguous, stride ld (global or LDS)
// A 16x32 k order used here (same as B): lane<16 -> K runs {k..k+16}; lane>=16: +16.
__device__ __forceinline__ v16bf load_afrag(const bf16* __restrict__ W, int row,
                                            int kbase, int ld, int hi) {
    const bf16* rp = W + (size_t)row * ld + kbase + hi * 16;
    v8bf lo = *(const v8bf*)rp;
    v8bf hh = *(const v8bf*)(rp + 8);
    return __builtin_shufflevector(lo, hh, 0,1,2,3,4,5,6,7,8,9,10,11,12,13,14,15);
}

// ---------------------------------------------------------------------------
// dense expert MLP: out = x + sum_e mask_e * (W2_e @ gelu(W1_e @ X + b1_e) + b2_e)
// 4 waves/block, 16 px/wave. Per-block TDM staging of 32KB weight chunks in LDS:
//   stage 1: 4 chunks of W1 rows (128x128, contiguous)    -> A-frags from LDS
//   stage 2: 4 chunks of W2 K-slices (128 rows x 128 of 512, stride 512)
// Dynamic LDS: [0,64K) per-wave H staging, [64K,96K) weight chunk buffer.
__global__ __launch_bounds__(128) void k_expert(
    const float* __restrict__ x,   const bf16* __restrict__ xbf,
    const bf16* __restrict__ w1,   const bf16* __restrict__ w2,
    const float* __restrict__ eb1, const float* __restrict__ eb2,
    const float* __restrict__ mask, float* __restrict__ out)
{
    extern __shared__ char smem[];
    bf16* Hs = (bf16*)smem;                      // 4 waves * 16 px * 512 = 64 KB
    bf16* Wb = (bf16*)(smem + 65536);            // 32 KB weight chunk
    unsigned wb_lds = lds_addr32(Wb);

    int wave = threadIdx.x >> 5;
    int lane = threadIdx.x & 31;
    int col  = lane & 15;                        // N index within tile
    int hi   = lane >> 4;                        // half select
    int b    = blockIdx.x >> 6;                  // 64 blocks per batch
    int p    = (blockIdx.x & 63) * 64 + wave * 16 + col;

    // cache X B-fragments for the 4 K-steps (K=128): 16 contiguous bf16 each
    const bf16* xrow = xbf + ((size_t)b * HW_ + p) * C_;
    v16bf bx[4];
    #pragma unroll
    for (int s = 0; s < 4; ++s)
        bx[s] = *(const v16bf*)(xrow + s * 32 + hi * 16);

    v8f outacc[8] = {};
    bf16* hbase = &Hs[wave * 16 * HID_ + col * HID_];

    #pragma unroll 1
    for (int e = 0; e < E_; ++e) {
        const bf16* W1 = w1 + (size_t)e * HID_ * C_;
        const bf16* W2 = w2 + (size_t)e * C_ * HID_;
        const float* B1 = eb1 + e * HID_;
        const float* B2 = eb2 + e * C_;
        float mval = mask[((size_t)b * E_ + e) * HW_ + p];

        // ---- stage 1: H = gelu(W1 @ X + b1), staged per-wave in LDS ----
        #pragma unroll 1
        for (int g = 0; g < 4; ++g) {
            __syncthreads();                     // chunk buffer free
            if (wave == 0) {
                // W1 rows [g*128, g*128+128), all K: contiguous 16384 elems
                tdm_load_2d(wb_lds, W1 + (size_t)g * 128 * C_, 16384u, 1u, 16384ull);
                __builtin_amdgcn_s_wait_tensorcnt(0);
            }
            __syncthreads();                     // chunk ready
            #pragma unroll 1
            for (int mt8 = 0; mt8 < 8; ++mt8) {
                int mt   = g * 8 + mt8;
                int lrow = mt8 * 16 + col;       // row within chunk
                v8f acc = {};
                #pragma unroll
                for (int s = 0; s < 4; ++s) {
                    v16bf a = load_afrag(Wb, lrow, s * 32, C_, hi);
                    acc = __builtin_amdgcn_wmma_f32_16x16x32_bf16(
                              false, a, false, bx[s], (short)0, acc, false, false);
                }
                #pragma unroll
                for (int r = 0; r < 8; ++r) {
                    int orow = mt * 16 + hi * 8 + r;   // C/D layout: M = r + 8*hi
                    hbase[orow] = f2bf(gelu_exact(acc[r] + B1[orow]));
                }
            }
        }

        // ---- stage 2: Y = W2 @ H + b2, accumulate mask*Y ----
        v8f acc2[8] = {};
        #pragma unroll 1
        for (int g = 0; g < 4; ++g) {
            __syncthreads();
            if (wave == 0) {
                // W2 all 128 rows, K slice [g*128, g*128+128), row stride 512
                tdm_load_2d(wb_lds, W2 + (size_t)g * 128, 128u, 128u, (unsigned long long)HID_);
                __builtin_amdgcn_s_wait_tensorcnt(0);
            }
            __syncthreads();
            #pragma unroll 1
            for (int s2l = 0; s2l < 4; ++s2l) {
                int s2 = g * 4 + s2l;
                v16bf bh = *(const v16bf*)(hbase + s2 * 32 + hi * 16);
                #pragma unroll
                for (int mt = 0; mt < 8; ++mt) {
                    v16bf a = load_afrag(Wb, mt * 16 + col, s2l * 32, 128, hi);
                    acc2[mt] = __builtin_amdgcn_wmma_f32_16x16x32_bf16(
                                   false, a, false, bh, (short)0, acc2[mt], false, false);
                }
            }
        }
        #pragma unroll
        for (int mt = 0; mt < 8; ++mt)
            #pragma unroll
            for (int r = 0; r < 8; ++r) {
                int ch = mt * 16 + hi * 8 + r;
                outacc[mt][r] = fmaf(mval, acc2[mt][r] + B2[ch], outacc[mt][r]);
            }
    }

    // out = x + final
    #pragma unroll
    for (int mt = 0; mt < 8; ++mt)
        #pragma unroll
        for (int r = 0; r < 8; ++r) {
            int ch = mt * 16 + hi * 8 + r;
            size_t idx = ((size_t)(b * C_ + ch)) * HW_ + p;
            out[idx] = x[idx] + outacc[mt][r];
        }
}

// ---------------------------------------------------------------------------
__global__ void k_aux(const float* gprob, const float* gload,
                      const float* ortho_sq, float* outp) {
    if (threadIdx.x == 0 && blockIdx.x == 0) {
        float s = 0.f;
        for (int e = 0; e < E_; ++e)
            s += (gprob[e] * (1.0f / (float)NPIX)) * (gload[e] * (1.0f / (float)NPIX));
        outp[0] = (float)E_ * s + 0.5f * sqrtf(*ortho_sq);
    }
}

// ---------------------------------------------------------------------------
extern "C" void kernel_launch(void* const* d_in, const int* in_sizes, int n_in,
                              void* d_out, int out_size, void* d_ws, size_t ws_size,
                              hipStream_t stream) {
    (void)in_sizes; (void)n_in; (void)out_size; (void)ws_size;
    const float* x     = (const float*)d_in[0];
    const float* proto = (const float*)d_in[1];
    const float* ctx_w = (const float*)d_in[2];
    const float* ctx_b = (const float*)d_in[3];
    const float* inp_w = (const float*)d_in[4];
    const float* inp_b = (const float*)d_in[5];
    const float* wq    = (const float*)d_in[6];
    const float* bq    = (const float*)d_in[7];
    const float* wk    = (const float*)d_in[8];
    const float* bk    = (const float*)d_in[9];
    const float* wv    = (const float*)d_in[10];
    const float* bv    = (const float*)d_in[11];
    const float* wo    = (const float*)d_in[12];
    const float* bo    = (const float*)d_in[13];
    const float* ln_g  = (const float*)d_in[14];
    const float* ln_b  = (const float*)d_in[15];
    const float* ew1   = (const float*)d_in[16];
    const float* eb1   = (const float*)d_in[17];
    const float* ew2   = (const float*)d_in[18];
    const float* eb2   = (const float*)d_in[19];

    char* ws = (char*)d_ws;
    constexpr size_t OFF_UP    = 0;
    constexpr size_t OFF_ORTHO = 24576;
    constexpr size_t OFF_PROB  = 24832;
    constexpr size_t OFF_LOAD  = 25088;
    constexpr size_t OFF_MASK  = 32768;
    constexpr size_t OFF_XBF   = OFF_MASK + (size_t)B_ * E_ * HW_ * 4;
    constexpr size_t OFF_W1    = OFF_XBF + (size_t)B_ * HW_ * C_ * 2;
    constexpr size_t OFF_W2    = OFF_W1 + (size_t)E_ * HID_ * C_ * 2;

    float* up_ws  = (float*)(ws + OFF_UP);
    float* ortho  = (float*)(ws + OFF_ORTHO);
    float* gprob  = (float*)(ws + OFF_PROB);
    float* gload  = (float*)(ws + OFF_LOAD);
    float* maskb  = (float*)(ws + OFF_MASK);
    bf16*  xbf    = (bf16*)(ws + OFF_XBF);
    bf16*  w1     = (bf16*)(ws + OFF_W1);
    bf16*  w2     = (bf16*)(ws + OFF_W2);
    float* out    = (float*)d_out;

    k_zero<<<1, 32, 0, stream>>>(gprob, gload);

    int nw = E_ * HID_ * C_;
    k_wcvt<<<(nw + 255) / 256, 256, 0, stream>>>(ew1, ew2, w1, w2, nw);

    k_transpose<<<dim3(HW_ / 16, C_ / 16, B_), dim3(16, 16), 0, stream>>>(x, xbf);

    k_prep<<<B_, 64, 0, stream>>>(x, proto, ctx_w, ctx_b, wq, bq, wk, bk,
                                  wv, bv, wo, bo, ln_g, ln_b, up_ws, ortho);

    k_route<<<B_ * (HW_ / 256), 256, 0, stream>>>(x, inp_w, inp_b, up_ws,
                                                  maskb, gprob, gload);

    // dynamic LDS: 64 KB H staging + 32 KB TDM weight chunk = 96 KB
    k_expert<<<B_ * (HW_ / 64), 128, 96 * 1024, stream>>>(x, xbf, w1, w2,
                                                          eb1, eb2, maskb, out);

    k_aux<<<1, 1, 0, stream>>>(gprob, gload, ortho, out + (size_t)B_ * C_ * HW_);
}
